// CompoundWordAutoregressiveWrapper_68135361184529
// MI455X (gfx1250) — compile-verified, matrix operand order checked by reading, not run on hardware
//
#include <hip/hip_runtime.h>
#include <hip/hip_bf16.h>
#include <math.h>

#define BT 8192          // B*T = 4*2048
#define VB 6913          // barbeat vocab
#define NG 12            // fourier bucket count / output dim

typedef __attribute__((ext_vector_type(2)))  float    v2f;
typedef __attribute__((ext_vector_type(8)))  float    v8f;
typedef __attribute__((ext_vector_type(16))) _Float16 v16h;

__device__ __forceinline__ float wredMax(float v) {
#pragma unroll
  for (int o = 16; o > 0; o >>= 1) v = fmaxf(v, __shfl_xor(v, o, 32));
  return v;
}
__device__ __forceinline__ float wredSum(float v) {
#pragma unroll
  for (int o = 16; o > 0; o >>= 1) v += __shfl_xor(v, o, 32);
  return v;
}

// basis[g][n]: g = bucket 0..11 (== i_val), n = output dim 0..11 (pad to 16 with 0)
// interleaved [sin(m*a), cos(m*a)] for m=1..6, a = -pi/6 * g.
// Branchless: cos(x) = sin(x + pi/2), selected via cndmask (keeps EXEC intact
// around the WMMAs instead of saveexec ladders).
__device__ __forceinline__ float basis_val(int g, int n) {
  float a     = -0.52359877559829887f * (float)g;       // -pi/6 * g
  float m     = (float)((n >> 1) + 1);
  float phase = (n & 1) ? 1.57079632679489662f : 0.0f;  // v_cndmask
  float v     = __sinf(__fmaf_rn(m, a, phase));
  return (n < NG) ? v : 0.0f;                           // v_cndmask
}

__global__ void __launch_bounds__(32) init_kernel(float* accum) {
  if (threadIdx.x < 8) accum[threadIdx.x] = 0.0f;   // 7 loss accums + mask count
}

// One block (256 thr = 8 waves) per row of barbeat [BT, 6913].
// Stage row in LDS, one streaming global read; compute max, sumexp,
// 12 bucket sums, CE term; write normalized buckets to `partial`.
__global__ void __launch_bounds__(256) barbeat_kernel(const float* __restrict__ logits,
                                                      const int*   __restrict__ target,
                                                      float* __restrict__ partial,
                                                      float* __restrict__ accum) {
  __shared__ float s[VB];
  __shared__ float red[8][4];
  const int row  = blockIdx.x;
  const int tid  = threadIdx.x;
  const int lane = tid & 31;
  const int wave = tid >> 5;
  const float* rowp = logits + (size_t)row * VB;

  float lmax = -3.402823466e38f;
  for (int i = tid; i < VB; i += 256) {
    __builtin_prefetch(rowp + i + 4096, 0, 1);   // global_prefetch_b8
    float v = rowp[i];
    s[i] = v;
    lmax = fmaxf(lmax, v);
  }
  lmax = wredMax(lmax);
  if (lane == 0) red[wave][0] = lmax;
  __syncthreads();
  float rmax = red[0][0];
#pragma unroll
  for (int w = 1; w < 8; ++w) rmax = fmaxf(rmax, red[w][0]);
  __syncthreads();

  // Buckets: prob index j+1, j = tid + 768*it (+0/+256/+512).
  // bucket(j) = (j>>6)%12 => static per thread: tid>>6, tid>>6+4, tid>>6+8.
  float a0 = 0.f, a1 = 0.f, a2 = 0.f;
#pragma unroll
  for (int it = 0; it < 9; ++it) {
    int j = tid + it * 768;
    a0 += __expf(s[j +   1] - rmax);
    a1 += __expf(s[j + 257] - rmax);
    a2 += __expf(s[j + 513] - rmax);
  }
  a0 = wredSum(a0); a1 = wredSum(a1); a2 = wredSum(a2);  // bucket set is wave-uniform
  if (lane == 0) {
    red[wave][0] = a0; red[wave][1] = a1; red[wave][2] = a2;
    red[wave][3] = a0 + a1 + a2;
  }
  __syncthreads();

  if (tid < 13) {
    float total = __expf(s[0] - rmax);           // index 0 belongs to softmax, not buckets
#pragma unroll
    for (int w = 0; w < 8; ++w) total += red[w][3];
    if (tid < 12) {
      int comp = tid >> 2;                       // which of a0/a1/a2
      int gg   = tid & 3;                        // which wave-pair (tid>>6 value)
      float gs = red[2 * gg][comp] + red[2 * gg + 1][comp];
      partial[(size_t)row * NG + tid] = gs / total;
    } else {
      int   tgt  = target[row * 7 + 1];
      float mask = (target[row * 7] != 0) ? 1.0f : 0.0f;
      float nll  = logf(total) + rmax - s[tgt];
      atomicAdd(accum + 1, mask * nll);          // barbeat loss slot
      atomicAdd(accum + 7, mask);                // mask count (accumulated once, here)
    }
  }
}

// One wave per (row, head) for the 6 small heads. 8192*6 = 49152 waves.
__global__ void __launch_bounds__(256) heads_kernel(
    const float* __restrict__ p0, const float* __restrict__ p2, const float* __restrict__ p3,
    const float* __restrict__ p4, const float* __restrict__ p5, const float* __restrict__ p6,
    const int* __restrict__ target, float* __restrict__ accum) {
  int gid    = blockIdx.x * 256 + threadIdx.x;
  int pairId = gid >> 5;                 // wave-uniform
  int lane   = gid & 31;
  int row    = pairId / 6;
  int h      = pairId - row * 6;
  const float* rowp; int Vh; int slot;
  switch (h) {
    case 0:  rowp = p0 + (size_t)row * 4;   Vh = 4;   slot = 0; break;
    case 1:  rowp = p2 + (size_t)row * 192; Vh = 192; slot = 2; break;
    case 2:  rowp = p3 + (size_t)row * 129; Vh = 129; slot = 3; break;
    case 3:  rowp = p4 + (size_t)row * 128; Vh = 128; slot = 4; break;
    case 4:  rowp = p5 + (size_t)row * 10;  Vh = 10;  slot = 5; break;
    default: rowp = p6 + (size_t)row * 64;  Vh = 64;  slot = 6; break;
  }
  float m = -3.402823466e38f;
  for (int i = lane; i < Vh; i += 32) m = fmaxf(m, rowp[i]);
  m = wredMax(m);
  int   tgt  = target[row * 7 + slot];
  float ssum = 0.0f, xt = 0.0f;
  for (int i = lane; i < Vh; i += 32) {
    float x = rowp[i];
    ssum += __expf(x - m);
    if (i == tgt) xt = x;                // exactly one lane captures it
  }
  ssum = wredSum(ssum);
  xt   = wredSum(xt);
  if (lane == 0) {
    float mask = (target[row * 7] != 0) ? 1.0f : 0.0f;
    float nll  = logf(ssum) + m - xt;
    atomicAdd(accum + slot, mask * nll);
  }
}

__global__ void __launch_bounds__(32) finalize_kernel(const float* __restrict__ accum,
                                                      float* __restrict__ out) {
  if (threadIdx.x < 7) out[threadIdx.x] = accum[threadIdx.x] / accum[7];
}

// d = partial[BT,12] x basis[12,12] via V_WMMA_F32_16X16X4_F32.
// One wave per 16-row tile; 3 K-steps of 4; N padded 12->16.
__global__ void __launch_bounds__(256) dmat_kernel(const float* __restrict__ partial,
                                                   float* __restrict__ dout) {
  const int lane    = threadIdx.x & 31;
  const int wave    = threadIdx.x >> 5;
  const int tile    = blockIdx.x * 8 + wave;   // 0..511, exact
  const int rowBase = tile * 16;
  const int half    = lane >> 4;               // lanes 16-31 hold K+2/K+3 (A), rows M+8 (D)
  const int l       = lane & 15;

  v8f c = {};
#if __has_builtin(__builtin_amdgcn_wmma_f32_16x16x4_f32)
#pragma unroll
  for (int k = 0; k < 12; k += 4) {
    // A 16x4 f32: lane<16 -> (M=l, K=k..k+1), lane>=16 -> (M=l, K=k+2..k+3)
    const float* pr = partial + (size_t)(rowBase + l) * NG + (k + half * 2);
    v2f a; a.x = pr[0]; a.y = pr[1];
    // B 4x16 f32: symmetric layout, N = l
    v2f b; b.x = basis_val(k + half * 2, l); b.y = basis_val(k + half * 2 + 1, l);
    c = __builtin_amdgcn_wmma_f32_16x16x4_f32(false, a, false, b, (short)0, c,
                                              false, false);
  }
#else
  // Fallback: codegen-confirmed f16 WMMA, K padded 12->32 with zeros.
  {
    const float* pr = partial + (size_t)(rowBase + l) * NG;
    v16h a, b;
#pragma unroll
    for (int e = 0; e < 16; ++e) {
      int v = e >> 1, hw = e & 1;
      int K = (half == 0) ? ((v < 4) ? (2 * v + hw) : (16 + 2 * (v - 4) + hw))
                          : ((v < 4) ? (8 + 2 * v + hw) : (24 + 2 * (v - 4) + hw));
      a[e] = (_Float16)((K < NG) ? pr[K] : 0.0f);
      b[e] = (_Float16)((K < NG) ? basis_val(K, l) : 0.0f);
    }
    c = __builtin_amdgcn_wmma_f32_16x16x32_f16(false, a, false, b, (short)0, c,
                                               false, false);
  }
#endif
#pragma unroll
  for (int v = 0; v < 8; ++v) {
    int row = rowBase + v + half * 8;          // C/D layout: VGPR v -> M=v / M=v+8
    if (l < NG) dout[(size_t)row * NG + l] = c[v];
  }
}

extern "C" void kernel_launch(void* const* d_in, const int* in_sizes, int n_in,
                              void* d_out, int out_size, void* d_ws, size_t ws_size,
                              hipStream_t stream) {
  const float* p_type  = (const float*)d_in[0];
  const float* p_bar   = (const float*)d_in[1];
  const float* p_tempo = (const float*)d_in[2];
  const float* p_instr = (const float*)d_in[3];
  const float* p_note  = (const float*)d_in[4];
  const float* p_oct   = (const float*)d_in[5];
  const float* p_dur   = (const float*)d_in[6];
  const int*   target  = (const int*)d_in[7];
  float* out = (float*)d_out;

  float* accum   = (float*)d_ws;          // 8 floats: loss accum[7] + mask count
  float* partial = (float*)d_ws + 16;     // [BT, 12] normalized bucket sums

  init_kernel<<<1, 32, 0, stream>>>(accum);
  barbeat_kernel<<<BT, 256, 0, stream>>>(p_bar, target, partial, accum);
  heads_kernel<<<(BT * 6 * 32) / 256, 256, 0, stream>>>(p_type, p_tempo, p_instr,
                                                        p_note, p_oct, p_dur,
                                                        target, accum);
  finalize_kernel<<<1, 32, 0, stream>>>(accum, out);          // out[0..6] = losses
  dmat_kernel<<<BT / 16 / 8, 256, 0, stream>>>(partial, out + 7);  // out[7..] = d
}
